// GINEScoreModel_23029614641878
// MI455X (gfx1250) — matrix operand order repaired and checked by reading, not compiled, not run on hardware
//
#include <hip/hip_runtime.h>

#define N_NODES 4096
#define NPG     512     // nodes per graph
#define NGRAPH  8
#define KNB     30      // real neighbors
#define KPAD    32      // padded edge rows per node (WMMA M tile x2)
#define NBASIS  32
#define EDIM    35      // NB + 3
#define EK      64      // padded edge feature K for WMMA
#define HD      256     // hidden
#define X2      512     // 2*H
#define TDIM    128
#define LAYERS  5

typedef __attribute__((ext_vector_type(16))) _Float16 v16h;
typedef __attribute__((ext_vector_type(8)))  _Float16 v8h;
typedef __attribute__((ext_vector_type(8)))  float    v8f;

static __device__ __forceinline__ float silu_f(float v) { return v / (1.f + __expf(-v)); }

// A-matrix 16x32 f16 fragment (ISA 7.12.2): lane r=l&15 holds row r;
// hi half-wave selects K-subblocks {0..7,16..23} vs {8..15,24..31}.
// 32-bit element offset against a uniform base -> saddr-form global loads.
static __device__ __forceinline__ v16h load_fragA(const _Float16* base, unsigned off, int ld) {
    int lane = threadIdx.x & 31;
    unsigned o = off + (unsigned)((lane & 15) * ld + (lane >> 4) * 8);
    v8h a = *(const v8h*)(base + o);
    v8h b = *(const v8h*)(base + o + 16);
    v16h f;
#pragma unroll
    for (int i = 0; i < 8; i++) { f[i] = a[i]; f[i + 8] = b[i]; }
    return f;
}

// B-matrix 32x16 f16 fragment from transposed (N-major) storage: lane c=l&15
// holds column c; lanes 0-15 hold K={0..7,16..23}, lanes 16-31 hold K={8..15,24..31}.
static __device__ __forceinline__ v16h load_fragB(const _Float16* baseT, unsigned off, int ld) {
    int lane = threadIdx.x & 31;
    unsigned o = off + (unsigned)((lane & 15) * ld + (lane >> 4) * 8);
    v8h a = *(const v8h*)(baseT + o);
    v8h b = *(const v8h*)(baseT + o + 16);
    v16h f;
#pragma unroll
    for (int i = 0; i < 8; i++) { f[i] = a[i]; f[i + 8] = b[i]; }
    return f;
}

static __device__ __forceinline__ v8f wmma_f16(v16h a, v16h b, v8f c) {
    return __builtin_amdgcn_wmma_f32_16x16x32_f16(false, a, false, b, (short)0, c, false, false);
}

static __device__ __forceinline__ void wait_async0() {
#if __has_builtin(__builtin_amdgcn_s_wait_asynccnt)
    __builtin_amdgcn_s_wait_asynccnt(0);
#else
    asm volatile("s_wait_asynccnt 0x0" ::: "memory");
#endif
}

// ---------------- prep kernels ----------------

__global__ void time_embed_kernel(const float* t, const float* W_gfp,
                                  const float* W_t, const float* b_t, float* tfeat) {
    __shared__ float feat[TDIM];
    int b = blockIdx.x, tid = threadIdx.x;
    if (tid < TDIM / 2) {
        float ang = 2.f * 3.14159265358979323846f * t[b] * W_gfp[tid];
        feat[tid] = sinf(ang);
        feat[tid + TDIM / 2] = cosf(ang);
    }
    __syncthreads();
    float acc = b_t[tid];
    for (int j = 0; j < TDIM; j++) acc += feat[j] * W_t[j * TDIM + tid];
    tfeat[b * TDIM + tid] = silu_f(acc);
}

__global__ void tpn_kernel(const float* tfeat, const float* Wt, const float* bt, float* tpn) {
    int l = blockIdx.x / NGRAPH, b = blockIdx.x % NGRAPH, f = threadIdx.x;
    float acc = bt[l * HD + f];
    for (int j = 0; j < TDIM; j++) acc += tfeat[b * TDIM + j] * Wt[((size_t)l * TDIM + j) * HD + f];
    tpn[((size_t)l * NGRAPH + b) * HD + f] = acc;
}

__global__ void node_embed_kernel(const float* coords, const float* W_in,
                                  const float* b_in, float* h) {
    int idx = blockIdx.x * blockDim.x + threadIdx.x;
    if (idx >= N_NODES * HD) return;
    int i = idx / HD, f = idx % HD;
    float acc = b_in[f] + coords[3 * i + 0] * W_in[0 * HD + f]
                        + coords[3 * i + 1] * W_in[1 * HD + f]
                        + coords[3 * i + 2] * W_in[2 * HD + f];
    h[idx] = acc;
}

__global__ void topk_kernel(const float* coords, const int* batch, int* src, float* maskb) {
    int i = blockIdx.x * blockDim.x + threadIdx.x;
    if (i >= N_NODES) return;
    float xi = coords[3 * i], yi = coords[3 * i + 1], zi = coords[3 * i + 2];
    int g = batch[i];
    int base = g * NPG;
    float bd[KNB]; int bi[KNB];
    for (int k = 0; k < KNB; k++) { bd[k] = 1e30f; bi[k] = 0; }
    for (int j = base; j < base + NPG; j++) {
        if (j == i) continue;
        float dx = coords[3 * j] - xi, dy = coords[3 * j + 1] - yi, dz = coords[3 * j + 2] - zi;
        float d = sqrtf(fmaxf(dx * dx + dy * dy + dz * dz, 1e-12f));
        if (d < 1.5f && d < bd[KNB - 1]) {
            int p = KNB - 1;
            while (p > 0 && bd[p - 1] > d) { bd[p] = bd[p - 1]; bi[p] = bi[p - 1]; p--; }
            bd[p] = d; bi[p] = j;
        }
    }
    for (int k = 0; k < KPAD; k++) {
        int s = 0; float m = 0.f;
        if (k < KNB && bd[k] < 1e29f) { s = bi[k]; m = 1.f; }
        src[i * KPAD + k] = s;
        maskb[i * KPAD + k] = m;
    }
}

// build padded f16 edge features: Aedge[node][32 rows][64 cols]
__global__ void edge_attr_kernel(const float* coords, const int* src, const float* maskb,
                                 _Float16* Aedge) {
    __shared__ float s_len[KPAD];
    __shared__ float s_dir[KPAD][3];
    int i = blockIdx.x, tid = threadIdx.x;
    if (tid < KPAD) {
        int s = src[i * KPAD + tid];
        float dx = coords[3 * s] - coords[3 * i];
        float dy = coords[3 * s + 1] - coords[3 * i + 1];
        float dz = coords[3 * s + 2] - coords[3 * i + 2];
        float len = sqrtf(fmaxf(dx * dx + dy * dy + dz * dz, 1e-12f));
        s_len[tid] = len;
        s_dir[tid][0] = dx / len; s_dir[tid][1] = dy / len; s_dir[tid][2] = dz / len;
    }
    __syncthreads();
    const float step = 1.5f / 33.f;
    for (int idx = tid; idx < KPAD * EK; idx += 256) {
        int e = idx >> 6, j = idx & 63;
        float v = 0.f;
        if (j < NBASIS) {
            float u = (s_len[e] - (float)(j + 1) * step) / step;
            v = __expf(-u * u) * (1.f / 1.12f);
        } else if (j < EDIM) {
            v = s_dir[e][j - NBASIS];
        }
        Aedge[(size_t)i * KPAD * EK + idx] = (_Float16)v;
    }
}

// f16 transposed weight prep
__global__ void conv_WeT_kernel(const float* We, _Float16* WeT) {
    int idx = blockIdx.x * 256 + threadIdx.x;
    if (idx >= LAYERS * X2 * EK) return;
    int l = idx / (X2 * EK), rem = idx % (X2 * EK);
    int n = rem / EK, k = rem % EK;
    float v = (k < EDIM) ? We[((size_t)l * EDIM + k) * X2 + n] : 0.f;
    WeT[idx] = (_Float16)v;
}
__global__ void conv_W1T_kernel(const float* W1, _Float16* W1T) {
    int idx = blockIdx.x * 256 + threadIdx.x;
    if (idx >= LAYERS * HD * X2) return;
    int l = idx / (HD * X2), rem = idx % (HD * X2);
    int n = rem / X2, k = rem % X2;
    W1T[idx] = (_Float16)W1[((size_t)l * X2 + k) * HD + n];
}
__global__ void conv_W2T_kernel(const float* W2, _Float16* W2T) {
    int idx = blockIdx.x * 256 + threadIdx.x;
    if (idx >= LAYERS * HD * HD) return;
    int l = idx / (HD * HD), rem = idx % (HD * HD);
    int n = rem / HD, k = rem % HD;
    W2T[idx] = (_Float16)W2[((size_t)l * HD + k) * HD + n];
}

// ---------------- per-layer kernels ----------------

__global__ void build_x_kernel(const float* h, const float* tpn_l, const int* batch, float* x) {
    int idx = blockIdx.x * 256 + threadIdx.x; // N*512 total
    int i = idx >> 9, f = idx & 511;
    x[idx] = (f < HD) ? h[(unsigned)(i * HD + f)] : tpn_l[(unsigned)(batch[i] * HD + (f - HD))];
}

// Fused: ea = eattr@We + be  ->  msg = relu(x[src]+ea)*mask  ->  aggr over K
//        -> z0 = (1+eps)*x + aggr, written f16 for the next WMMA GEMM.
// Edge tile staged into LDS with CDNA5 async global->LDS loads (ASYNCcnt).
__global__ __launch_bounds__(256) void edge_layer_kernel(
    const _Float16* __restrict__ Aedge, const _Float16* __restrict__ WeT,
    const float* __restrict__ be, const int* __restrict__ src,
    const float* __restrict__ maskb, const float* __restrict__ x,
    const float* __restrict__ eps, int l, _Float16* __restrict__ Z0h) {
    __shared__ __align__(16) _Float16 Alds[KPAD * EK];
    __shared__ int   s_soff[KPAD];   // src * X2 (precomputed 32-bit gather offsets)
    __shared__ float s_msk[KPAD];
    int i = blockIdx.x, tid = threadIdx.x;

    // async copy of the 4KB edge-feature tile: 256 threads x 16B, no VGPR staging
    {
        const _Float16* g = Aedge + (size_t)i * KPAD * EK + tid * 8;
        unsigned lds_off = (unsigned)(size_t)(const void*)(Alds + tid * 8);
        asm volatile("global_load_async_to_lds_b128 %0, %1, off"
                     :: "v"(lds_off), "v"(g) : "memory");
    }
    if (tid < KPAD) {
        s_soff[tid] = src[i * KPAD + tid] * X2;
        s_msk[tid]  = maskb[i * KPAD + tid];
    }
    wait_async0();
    __syncthreads();

    int wave = tid >> 5, lane = tid & 31, col = lane & 15, hi = lane >> 4;
    int nbase = wave * 64;                 // 8 waves x 64 cols = 512
    v8f acc[2][4] = {};
#pragma unroll
    for (int k0 = 0; k0 < EK; k0 += 32) {
        v16h a0 = load_fragA(Alds, (unsigned)k0, EK);
        v16h a1 = load_fragA(Alds, (unsigned)(16 * EK + k0), EK);
#pragma unroll
        for (int nt = 0; nt < 4; nt++) {
            v16h bfr = load_fragB(WeT, (unsigned)((nbase + nt * 16) * EK + k0), EK);
            acc[0][nt] = wmma_f16(a0, bfr, acc[0][nt]);
            acc[1][nt] = wmma_f16(a1, bfr, acc[1][nt]);
        }
    }
    float epsl = 1.f + eps[l];
#pragma unroll
    for (int nt = 0; nt < 4; nt++) {
        int n = nbase + nt * 16 + col;
        float bv = be[n];
        float sum = 0.f;
#pragma unroll
        for (int mt = 0; mt < 2; mt++) {
#pragma unroll
            for (int r = 0; r < 8; r++) {
                int e = mt * 16 + 8 * hi + r;   // edge row this lane/vgpr holds
                float v = acc[mt][nt][r] + bv + x[(unsigned)(s_soff[e] + n)];
                sum += fmaxf(v, 0.f) * s_msk[e];
            }
        }
        sum += __shfl_xor(sum, 16);             // lanes l and l^16 share column n
        if (hi == 0) {
            float z0 = epsl * x[(unsigned)(i * X2 + n)] + sum;
            Z0h[(unsigned)(i * X2 + n)] = (_Float16)z0;
        }
    }
}

// Generic f16 WMMA GEMM: C[M,Nn] = act(A[M,Kd] @ Bt[Nn,Kd]^T + bias)
// block = 4 waves, tile 64(M) x 64(N)
template <int ACT, int OUTH>
__global__ __launch_bounds__(128) void gemm_wmma_kernel(
    const _Float16* __restrict__ A, const _Float16* __restrict__ Bt,
    const float* __restrict__ bias, float* __restrict__ Cf,
    _Float16* __restrict__ Ch, int Nn, int Kd) {
    int wave = threadIdx.x >> 5, lane = threadIdx.x & 31;
    int col = lane & 15, hi = lane >> 4;
    int m0 = blockIdx.x * 64 + wave * 16;
    int n0 = blockIdx.y * 64;
    v8f acc[4] = {};
    for (int k0 = 0; k0 < Kd; k0 += 32) {
        v16h a = load_fragA(A, (unsigned)(m0 * Kd + k0), Kd);
#pragma unroll
        for (int nt = 0; nt < 4; nt++) {
            v16h b = load_fragB(Bt, (unsigned)((n0 + nt * 16) * Kd + k0), Kd);
            acc[nt] = wmma_f16(a, b, acc[nt]);
        }
    }
#pragma unroll
    for (int nt = 0; nt < 4; nt++) {
        int n = n0 + nt * 16 + col;
        float bv = bias[n];
#pragma unroll
        for (int r = 0; r < 8; r++) {
            int m = m0 + 8 * hi + r;
            float v = acc[nt][r] + bv;
            if (ACT == 1) v = silu_f(v);
            unsigned idx = (unsigned)(m * Nn + n);
            if (OUTH) Ch[idx] = (_Float16)v;
            else      Cf[idx] = v;
        }
    }
}

__global__ void graphnorm_kernel(const float* Z2, const float* gn_w, const float* gn_b,
                                 const float* gn_ms, float* h) {
    int g = blockIdx.x, f = threadIdx.x;
    unsigned base = (unsigned)(g * NPG * HD + f);
    float mean = 0.f;
    for (int n = 0; n < NPG; n++) mean += Z2[base + (unsigned)n * HD];
    mean *= (1.f / NPG);
    float ms = mean * gn_ms[f];
    float var = 0.f;
    for (int n = 0; n < NPG; n++) { float c = Z2[base + (unsigned)n * HD] - ms; var += c * c; }
    var *= (1.f / NPG);
    float inv = rsqrtf(var + 1e-5f);
    float w = gn_w[f], bb = gn_b[f];
    for (int n = 0; n < NPG; n++) {
        unsigned idx = base + (unsigned)n * HD;
        float c = Z2[idx] - ms;
        float s = w * c * inv + bb + h[idx];
        h[idx] = silu_f(s);
    }
}

__global__ void out_proj_kernel(const float* h, const float* W_out, const float* b_out,
                                float* out) {
    int idx = blockIdx.x * blockDim.x + threadIdx.x;
    if (idx >= N_NODES * 3) return;
    int i = idx / 3, c = idx % 3;
    float acc = b_out[c];
    for (int f = 0; f < HD; f++) acc += h[(unsigned)(i * HD + f)] * W_out[f * 3 + c];
    out[idx] = acc;
}

// ---------------- host launch ----------------

extern "C" void kernel_launch(void* const* d_in, const int* in_sizes, int n_in,
                              void* d_out, int out_size, void* d_ws, size_t ws_size,
                              hipStream_t stream) {
    const float* coords = (const float*)d_in[0];
    const int*   batch  = (const int*)d_in[1];
    const float* t      = (const float*)d_in[2];
    const float* W_gfp  = (const float*)d_in[3];
    const float* W_t    = (const float*)d_in[4];
    const float* b_t    = (const float*)d_in[5];
    const float* W_in   = (const float*)d_in[6];
    const float* b_in   = (const float*)d_in[7];
    const float* W1     = (const float*)d_in[8];
    const float* b1     = (const float*)d_in[9];
    const float* W2     = (const float*)d_in[10];
    const float* b2     = (const float*)d_in[11];
    const float* We     = (const float*)d_in[12];
    const float* be     = (const float*)d_in[13];
    const float* eps    = (const float*)d_in[14];
    const float* gn_w   = (const float*)d_in[15];
    const float* gn_b   = (const float*)d_in[16];
    const float* gn_ms  = (const float*)d_in[17];
    const float* Wt     = (const float*)d_in[18];
    const float* bt     = (const float*)d_in[19];
    const float* W_out  = (const float*)d_in[20];
    const float* b_out  = (const float*)d_in[21];
    float* out = (float*)d_out;

    size_t off = 0;
    char* wsb = (char*)d_ws;
    auto take = [&](size_t bytes) -> char* {
        char* p = wsb + off;
        off += (bytes + 255) & ~(size_t)255;
        return p;
    };
    float*     tfeat = (float*)take((size_t)NGRAPH * TDIM * 4);
    float*     tpn   = (float*)take((size_t)LAYERS * NGRAPH * HD * 4);
    int*       srcb  = (int*)take((size_t)N_NODES * KPAD * 4);
    float*     maskb = (float*)take((size_t)N_NODES * KPAD * 4);
    _Float16*  Aedge = (_Float16*)take((size_t)N_NODES * KPAD * EK * 2);
    _Float16*  WeT   = (_Float16*)take((size_t)LAYERS * X2 * EK * 2);
    _Float16*  W1T   = (_Float16*)take((size_t)LAYERS * HD * X2 * 2);
    _Float16*  W2T   = (_Float16*)take((size_t)LAYERS * HD * HD * 2);
    float*     hbuf  = (float*)take((size_t)N_NODES * HD * 4);
    float*     xbuf  = (float*)take((size_t)N_NODES * X2 * 4);
    _Float16*  Z0h   = (_Float16*)take((size_t)N_NODES * X2 * 2);
    _Float16*  Z1h   = (_Float16*)take((size_t)N_NODES * HD * 2);
    float*     Z2    = (float*)take((size_t)N_NODES * HD * 4);

    // prep
    time_embed_kernel<<<NGRAPH, TDIM, 0, stream>>>(t, W_gfp, W_t, b_t, tfeat);
    tpn_kernel<<<LAYERS * NGRAPH, HD, 0, stream>>>(tfeat, Wt, bt, tpn);
    node_embed_kernel<<<(N_NODES * HD + 255) / 256, 256, 0, stream>>>(coords, W_in, b_in, hbuf);
    topk_kernel<<<N_NODES / 256, 256, 0, stream>>>(coords, batch, srcb, maskb);
    edge_attr_kernel<<<N_NODES, 256, 0, stream>>>(coords, srcb, maskb, Aedge);
    conv_WeT_kernel<<<(LAYERS * X2 * EK + 255) / 256, 256, 0, stream>>>(We, WeT);
    conv_W1T_kernel<<<(LAYERS * HD * X2 + 255) / 256, 256, 0, stream>>>(W1, W1T);
    conv_W2T_kernel<<<(LAYERS * HD * HD + 255) / 256, 256, 0, stream>>>(W2, W2T);

    for (int l = 0; l < LAYERS; l++) {
        build_x_kernel<<<(N_NODES * X2) / 256, 256, 0, stream>>>(
            hbuf, tpn + (size_t)l * NGRAPH * HD, batch, xbuf);
        edge_layer_kernel<<<N_NODES, 256, 0, stream>>>(
            Aedge, WeT + (size_t)l * X2 * EK, be + (size_t)l * X2,
            srcb, maskb, xbuf, eps, l, Z0h);
        gemm_wmma_kernel<1, 1><<<dim3(N_NODES / 64, HD / 64), 128, 0, stream>>>(
            Z0h, W1T + (size_t)l * HD * X2, b1 + (size_t)l * HD, nullptr, Z1h, HD, X2);
        gemm_wmma_kernel<0, 0><<<dim3(N_NODES / 64, HD / 64), 128, 0, stream>>>(
            Z1h, W2T + (size_t)l * HD * HD, b2 + (size_t)l * HD, Z2, nullptr, HD, HD);
        graphnorm_kernel<<<NGRAPH, HD, 0, stream>>>(
            Z2, gn_w + (size_t)l * HD, gn_b + (size_t)l * HD, gn_ms + (size_t)l * HD, hbuf);
    }

    out_proj_kernel<<<(N_NODES * 3 + 255) / 256, 256, 0, stream>>>(hbuf, W_out, b_out, out);
    (void)in_sizes; (void)n_in; (void)out_size; (void)ws_size;
}
